// _Memory_Block_85564338471488
// MI455X (gfx1250) — matrix operand order repaired and checked by reading, not compiled, not run on hardware
//
#include <hip/hip_runtime.h>
#include <hip/hip_bf16.h>
#include <math.h>

typedef _Float16 half_t;
typedef __attribute__((ext_vector_type(16))) _Float16 v16h;
typedef __attribute__((ext_vector_type(8)))  _Float16 v8h;
typedef __attribute__((ext_vector_type(8)))  float    v8f;

union Frag16 { v16h v; v8h h[2]; };

#define RATE 0.999f
#define EPSC 1e-6f

// ---- problem dims ----
#define NROWS 65536      // b*h*w = 16*64*64
#define CDIM  256
#define KDIM  512
#define HW    4096       // 64*64
#define PLANE 1048576    // 256*4096 floats per batch image

// ---- ws byte offsets ----
#define WS_RNORM  0          // 65536 f32
#define WS_IND    262144     // 65536 i32
#define WS_COUNTS 524288     // 512 f32
#define WS_ESUM   526336     // 512*256 f32 (contiguous after counts)
#define WS_MN     1050624    // 512*256 f16  (l2norm(units))
#define WS_MHAT   1312768    // 512*256 f16  (l2norm(m))
#define WS_MT     1574912    // 256*512 f16  (m transposed)

// A-operand fragment (16x32 f16, MxK): lane<16 -> row=lane, K={kb..kb+7, kb+16..kb+23}
//                                      lane>=16 -> row=lane-16, K={kb+8..kb+15, kb+24..kb+31}
__device__ __forceinline__ v16h load_a_frag(const half_t* base, int stride, int lane, int kb) {
  int row = lane & 15;
  int ks  = kb + ((lane >> 4) << 3);
  Frag16 f;
  f.h[0] = *(const v8h*)(base + row * stride + ks);
  f.h[1] = *(const v8h*)(base + row * stride + ks + 16);
  return f.v;
}
// B-operand fragment (32x16 f16, KxN): lane<16 -> col=lane, K={kb..kb+7, kb+16..kb+23} ;
//                                      lane>=16 -> col=lane-16, K={kb+8..kb+15, kb+24..kb+31}
// "base" points at the N-major source (row index == N/column of B).
__device__ __forceinline__ v16h load_b_frag(const half_t* base, int stride, int lane, int kb) {
  int row = lane & 15;
  int ks  = kb + ((lane >> 4) << 3);
  Frag16 f;
  f.h[0] = *(const v8h*)(base + row * stride + ks);
  f.h[1] = *(const v8h*)(base + row * stride + ks + 16);
  return f.v;
}

__device__ __forceinline__ v8f wmma_f16(v16h a, v16h b, v8f c) {
  return __builtin_amdgcn_wmma_f32_16x16x32_f16(false, a, false, b, (short)0, c, false, false);
}

// ---------------- small utility kernels ----------------

__global__ __launch_bounds__(256) void k_zero(float* __restrict__ p, int n) {
  int i = blockIdx.x * 256 + threadIdx.x;
  if (i < n) p[i] = 0.0f;
}

// reciprocal L2 norms of the 65536 rows of x (strided gather over c)
__global__ __launch_bounds__(256) void k_rnorm(const float* __restrict__ x,
                                               float* __restrict__ rnorm) {
  int n = blockIdx.x * 256 + threadIdx.x;
  const float* xb = x + ((size_t)(n >> 12) << 20) + (n & 4095);
  float s = 0.0f;
#pragma unroll 8
  for (int c = 0; c < CDIM; ++c) {
    float v = xb[(size_t)c << 12];
    s += v * v;
  }
  rnorm[n] = 1.0f / fmaxf(sqrtf(s), 1e-12f);
}

// mn = l2norm(units) as f16, row-major (k,c)
__global__ __launch_bounds__(256) void k_prep_mn(const float* __restrict__ units,
                                                 half_t* __restrict__ mn) {
  __shared__ float sred[256];
  int k = blockIdx.x, c = threadIdx.x;
  float v = units[(size_t)k * CDIM + c];
  sred[c] = v * v;
  __syncthreads();
  for (int s = 128; s > 0; s >>= 1) {
    if (c < s) sred[c] += sred[c + s];
    __syncthreads();
  }
  float rinv = 1.0f / fmaxf(sqrtf(sred[0]), 1e-12f);
  mn[(size_t)k * CDIM + c] = (half_t)(v * rinv);
}

// ---------------- GEMM1 + fused argmax ----------------
// Each WG: 64 rows x all 512 cols, K=256. 8 waves, wave w owns cols [64w,64w+64).
__global__ __launch_bounds__(256)
void k_gemm_argmax(const float* __restrict__ x, const float* __restrict__ rnorm,
                   const half_t* __restrict__ mn, int* __restrict__ embed_ind) {
  __shared__ __align__(16) half_t Asm[64 * 264];
  __shared__ float wv[8][64];
  __shared__ int   wi[8][64];

  const int tid = threadIdx.x;
  const int lane = tid & 31;
  const int wave = tid >> 5;
  const int rowBase = blockIdx.x << 6;

  // stage normalized x rows as f16 (coalesced: fixed c, consecutive rows)
  {
    int r = tid & 63, cg = tid >> 6;
    int n = rowBase + r;
    const float* xb = x + ((size_t)(n >> 12) << 20) + (n & 4095);
    float rn = rnorm[n];
#pragma unroll 4
    for (int c = cg * 64; c < cg * 64 + 64; ++c)
      Asm[r * 264 + c] = (half_t)(xb[(size_t)c << 12] * rn);
  }
  __syncthreads();

  const int colBase = wave << 6;
  v8f zero = {0.f, 0.f, 0.f, 0.f, 0.f, 0.f, 0.f, 0.f};
  v8f acc[4][4];
#pragma unroll
  for (int mt = 0; mt < 4; ++mt)
#pragma unroll
    for (int nt = 0; nt < 4; ++nt) acc[mt][nt] = zero;

#pragma unroll
  for (int kk = 0; kk < 256; kk += 32) {
    v16h a[4], b[4];
#pragma unroll
    for (int mt = 0; mt < 4; ++mt) a[mt] = load_a_frag(Asm + mt * 16 * 264, 264, lane, kk);
#pragma unroll
    for (int nt = 0; nt < 4; ++nt)
      b[nt] = load_b_frag(mn + (size_t)(colBase + nt * 16) * CDIM, CDIM, lane, kk);
#pragma unroll
    for (int mt = 0; mt < 4; ++mt)
#pragma unroll
      for (int nt = 0; nt < 4; ++nt) acc[mt][nt] = wmma_f16(a[mt], b[nt], acc[mt][nt]);
  }

  // per-wave argmax over its 64-column strip, then cross-wave via LDS
#pragma unroll
  for (int mt = 0; mt < 4; ++mt) {
    float bv[8]; int bi[8];
#pragma unroll
    for (int j = 0; j < 8; ++j) { bv[j] = acc[mt][0][j]; bi[j] = colBase + (lane & 15); }
#pragma unroll
    for (int nt = 1; nt < 4; ++nt) {
      int ix = colBase + nt * 16 + (lane & 15);
#pragma unroll
      for (int j = 0; j < 8; ++j) {
        float v = acc[mt][nt][j];
        if (v > bv[j]) { bv[j] = v; bi[j] = ix; }
      }
    }
#pragma unroll
    for (int j = 0; j < 8; ++j) {
      for (int off = 1; off < 16; off <<= 1) {
        float ov = __shfl_xor(bv[j], off, 32);
        int   oi = __shfl_xor(bi[j], off, 32);
        if (ov > bv[j] || (ov == bv[j] && oi < bi[j])) { bv[j] = ov; bi[j] = oi; }
      }
      if ((lane & 15) == 0) {
        int row = mt * 16 + j + ((lane >> 4) << 3);
        wv[wave][row] = bv[j];
        wi[wave][row] = bi[j];
      }
    }
  }
  __syncthreads();
  if (tid < 64) {
    float best = wv[0][tid]; int bidx = wi[0][tid];
#pragma unroll
    for (int w = 1; w < 8; ++w) {
      float v = wv[w][tid]; int ix = wi[w][tid];
      if (v > best || (v == best && ix < bidx)) { best = v; bidx = ix; }
    }
    embed_ind[rowBase + tid] = bidx;
  }
}

// ---------------- EMA scatter ----------------
__global__ __launch_bounds__(256)
void k_scatter(const float* __restrict__ x, const int* __restrict__ ind,
               float* __restrict__ esum, float* __restrict__ counts) {
  const int tid = threadIdx.x;
  const int rowBase = blockIdx.x << 6;
  const int r = tid & 63, cg = tid >> 6;
  const int n = rowBase + r;
  const int idx = ind[n];
  const float* xb = x + ((size_t)(n >> 12) << 20) + (n & 4095);
  float* dst = esum + (size_t)idx * CDIM;
#pragma unroll 4
  for (int c = cg * 64; c < cg * 64 + 64; ++c)
    atomicAdd(&dst[c], xb[(size_t)c << 12]);
  if (tid < 64) atomicAdd(&counts[idx], 1.0f);
}

// m = units*RATE + (esum/(counts+eps))*(1-RATE); emit l2norm(m) f16 and m^T f16
__global__ __launch_bounds__(256)
void k_finalize(const float* __restrict__ units, const float* __restrict__ esum,
                const float* __restrict__ counts, half_t* __restrict__ mhat,
                half_t* __restrict__ mT) {
  __shared__ float sred[256];
  int k = blockIdx.x, c = threadIdx.x;
  float cnt = counts[k] + EPSC;
  float m = units[(size_t)k * CDIM + c] * RATE + (esum[(size_t)k * CDIM + c] / cnt) * (1.0f - RATE);
  mT[(size_t)c * KDIM + k] = (half_t)m;
  sred[c] = m * m;
  __syncthreads();
  for (int s = 128; s > 0; s >>= 1) {
    if (c < s) sred[c] += sred[c + s];
    __syncthreads();
  }
  float rinv = 1.0f / fmaxf(sqrtf(sred[0]), 1e-12f);
  mhat[(size_t)k * CDIM + c] = (half_t)(m * rinv);
}

// ---------------- fused GEMM2 -> softmax -> GEMM3 ----------------
// Per WG: 64-row stripe. score2 (f32) -> d_out, probs kept in LDS, out -> d_out.
// Single aliased LDS buffer:
//   phase A/B: [0,33792) xn tile f16 (stride 264) | [33792,100352) score/prob tile
//              f16 (stride 520) | [100352,102400) softmax partials f32
//   epilogue : [0,69632) out tile f32, layout [ci][68] (272B stride, 16B aligned)
#define SM_ASM  0
#define SM_SSM  33792
#define SM_RED  100352
#define SM_SIZE 102400

__global__ __launch_bounds__(256)
void k_fused(const float* __restrict__ x, const float* __restrict__ rnorm,
             const half_t* __restrict__ mhat, const half_t* __restrict__ mT,
             float* __restrict__ out, float* __restrict__ score) {
  __shared__ __align__(16) char smem[SM_SIZE];
  half_t* Asm = (half_t*)(smem + SM_ASM);   // 64 x 264
  half_t* Ssm = (half_t*)(smem + SM_SSM);   // 64 x 520
  float*  red = (float*)(smem + SM_RED);    // 64 x 8

  const int tid = threadIdx.x;
  const int lane = tid & 31;
  const int wave = tid >> 5;
  const int rowBase = blockIdx.x << 6;

  {
    int r = tid & 63, cg = tid >> 6;
    int n = rowBase + r;
    const float* xb = x + ((size_t)(n >> 12) << 20) + (n & 4095);
    float rn = rnorm[n];
#pragma unroll 4
    for (int c = cg * 64; c < cg * 64 + 64; ++c)
      Asm[r * 264 + c] = (half_t)(xb[(size_t)c << 12] * rn);
  }
  __syncthreads();

  // ---- GEMM2: score2 = xn @ mhat^T ; wave owns cols [64w, 64w+64), K=256
  {
    const int colBase = wave << 6;
    v8f zero = {0.f, 0.f, 0.f, 0.f, 0.f, 0.f, 0.f, 0.f};
    v8f acc[4][4];
#pragma unroll
    for (int mt = 0; mt < 4; ++mt)
#pragma unroll
      for (int nt = 0; nt < 4; ++nt) acc[mt][nt] = zero;

#pragma unroll
    for (int kk = 0; kk < 256; kk += 32) {
      v16h a[4], b[4];
#pragma unroll
      for (int mt = 0; mt < 4; ++mt) a[mt] = load_a_frag(Asm + mt * 16 * 264, 264, lane, kk);
#pragma unroll
      for (int nt = 0; nt < 4; ++nt)
        b[nt] = load_b_frag(mhat + (size_t)(colBase + nt * 16) * CDIM, CDIM, lane, kk);
#pragma unroll
      for (int mt = 0; mt < 4; ++mt)
#pragma unroll
        for (int nt = 0; nt < 4; ++nt) acc[mt][nt] = wmma_f16(a[mt], b[nt], acc[mt][nt]);
    }
    // write f32 score to HBM (64B-coalesced per instruction), f16 copy to LDS
#pragma unroll
    for (int mt = 0; mt < 4; ++mt)
#pragma unroll
      for (int nt = 0; nt < 4; ++nt) {
        int N = colBase + nt * 16 + (lane & 15);
#pragma unroll
        for (int j = 0; j < 8; ++j) {
          int M = mt * 16 + j + ((lane >> 4) << 3);
          float v = acc[mt][nt][j];
          score[(size_t)(rowBase + M) * KDIM + N] = v;
          Ssm[M * 520 + N] = (half_t)v;
        }
      }
  }
  __syncthreads();

  // ---- softmax over 512 cols per row; thread (row, quarter) scans 128 elems
  {
    int row = tid >> 2, q = tid & 3;
    half_t* srow = Ssm + row * 520;
    float lm = -1e30f;
    for (int c = q * 128; c < q * 128 + 128; ++c) lm = fmaxf(lm, (float)srow[c]);
    red[row * 8 + q] = lm;
    __syncthreads();
    float mx = fmaxf(fmaxf(red[row * 8 + 0], red[row * 8 + 1]),
                     fmaxf(red[row * 8 + 2], red[row * 8 + 3]));
    float ls = 0.0f;
    for (int c = q * 128; c < q * 128 + 128; ++c) ls += __expf((float)srow[c] - mx);
    red[row * 8 + 4 + q] = ls;
    __syncthreads();
    float tot = red[row * 8 + 4] + red[row * 8 + 5] + red[row * 8 + 6] + red[row * 8 + 7];
    float rs = 1.0f / tot;
    for (int c = q * 128; c < q * 128 + 128; ++c)
      srow[c] = (half_t)(__expf((float)srow[c] - mx) * rs);
  }
  __syncthreads();

  // ---- GEMM3: out = P(64x512) @ m(512x256); wave owns cols [32w, 32w+32), K=512
  {
    const int colBase = wave << 5;
    v8f zero = {0.f, 0.f, 0.f, 0.f, 0.f, 0.f, 0.f, 0.f};
    v8f acc[4][2];
#pragma unroll
    for (int mt = 0; mt < 4; ++mt)
#pragma unroll
      for (int nt = 0; nt < 2; ++nt) acc[mt][nt] = zero;

    for (int kk = 0; kk < 512; kk += 32) {
      v16h a[4], b[2];
#pragma unroll
      for (int mt = 0; mt < 4; ++mt) a[mt] = load_a_frag(Ssm + mt * 16 * 520, 520, lane, kk);
#pragma unroll
      for (int nt = 0; nt < 2; ++nt)
        b[nt] = load_b_frag(mT + (size_t)(colBase + nt * 16) * KDIM, KDIM, lane, kk);
#pragma unroll
      for (int mt = 0; mt < 4; ++mt)
#pragma unroll
        for (int nt = 0; nt < 2; ++nt) acc[mt][nt] = wmma_f16(a[mt], b[nt], acc[mt][nt]);
    }

    // Stage out tile in LDS (reuse dead A/score storage), then coalesced b128 stores.
    __syncthreads();                      // all waves done reading Ssm
    float* Osm = (float*)smem;            // [ci][68] f32, conflict-free strides
#pragma unroll
    for (int mt = 0; mt < 4; ++mt)
#pragma unroll
      for (int nt = 0; nt < 2; ++nt) {
        int ci = colBase + nt * 16 + (lane & 15);
#pragma unroll
        for (int j = 0; j < 8; ++j) {
          int M = mt * 16 + j + ((lane >> 4) << 3);
          Osm[ci * 68 + M] = acc[mt][nt][j];
        }
      }
    __syncthreads();
    // thread (g = tid>>4 picks ci, t = tid&15 picks 4-row chunk): 16 lanes per ci
    // write one contiguous 256B run as 16 float4 stores, one per lane.
    {
      int g = tid >> 4, t = tid & 15;
      float* ob = out + ((size_t)(rowBase >> 12) << 20) + (rowBase & 4095) + t * 4;
#pragma unroll 4
      for (int it = 0; it < 16; ++it) {
        int ci = it * 16 + g;
        float4 v = *(const float4*)(Osm + ci * 68 + t * 4);
        *(float4*)(ob + ((size_t)ci << 12)) = v;
      }
    }
  }
}

// ---------------- launcher ----------------
extern "C" void kernel_launch(void* const* d_in, const int* in_sizes, int n_in,
                              void* d_out, int out_size, void* d_ws, size_t ws_size,
                              hipStream_t stream) {
  const float* x     = (const float*)d_in[0];
  const float* units = (const float*)d_in[1];
  (void)in_sizes; (void)n_in; (void)out_size; (void)ws_size;

  char* ws = (char*)d_ws;
  float*  rnorm  = (float*)(ws + WS_RNORM);
  int*    ind    = (int*)(ws + WS_IND);
  float*  counts = (float*)(ws + WS_COUNTS);
  float*  esum   = (float*)(ws + WS_ESUM);
  half_t* mn     = (half_t*)(ws + WS_MN);
  half_t* mhat   = (half_t*)(ws + WS_MHAT);
  half_t* mT     = (half_t*)(ws + WS_MT);

  float* out   = (float*)d_out;                 // 16*256*64*64
  float* score = (float*)d_out + 16777216;      // 65536*512

  // counts + esum are contiguous: 512 + 131072 floats
  k_zero<<<dim3((131584 + 255) / 256), dim3(256), 0, stream>>>(counts, 131584);
  k_rnorm<<<dim3(NROWS / 256), dim3(256), 0, stream>>>(x, rnorm);
  k_prep_mn<<<dim3(KDIM), dim3(256), 0, stream>>>(units, mn);
  k_gemm_argmax<<<dim3(NROWS / 64), dim3(256), 0, stream>>>(x, rnorm, mn, ind);
  k_scatter<<<dim3(NROWS / 64), dim3(256), 0, stream>>>(x, ind, esum, counts);
  k_finalize<<<dim3(KDIM), dim3(256), 0, stream>>>(units, esum, counts, mhat, mT);
  k_fused<<<dim3(NROWS / 64), dim3(256), 0, stream>>>(x, rnorm, mhat, mT, out, score);
}
